// UnpoolWithSkip_2508260901458
// MI455X (gfx1250) — matrix operand order repaired and checked by reading, not compiled, not run on hardware
//
#include <hip/hip_runtime.h>

typedef __attribute__((ext_vector_type(16))) __bf16 v16bf;
typedef __attribute__((ext_vector_type(8)))  __bf16 v8bf;
typedef __attribute__((ext_vector_type(8)))  float  v8f;

#define C_OUT 256

// LDS byte offset of a __shared__ object: low 32 bits of the flat
// shared-aperture address are the LDS offset (ISA 10.2 aperture mapping).
#define LDS_OFF(p) ((unsigned)(unsigned long long)(const void*)(p))

// ---------------------------------------------------------------------------
// Convert W [K][256] f32 (row-major) -> Wt [256][K] bf16 (transposed) so that
// WMMA B fragments (lane = column, K contiguous per half-wave) are linear loads.
// ---------------------------------------------------------------------------
__global__ __launch_bounds__(256) void convert_wt(const float* __restrict__ W,
                                                  __bf16* __restrict__ Wt,
                                                  int K, int total) {
    int idx = blockIdx.x * 256 + threadIdx.x;
    if (idx >= total) return;
    int n = idx / K;
    int k = idx - n * K;
    Wt[idx] = (__bf16)W[(size_t)k * C_OUT + n];
}

// ---------------------------------------------------------------------------
// GEMM with bf16 WMMA, f32 accumulate, async global->LDS double buffering.
//   C[rowBase..+64][colBase..+128] = A @ Wt^T + bias
// Also emits deterministic per-workgroup column partial sums / sum-of-squares
// for the BatchNorm batch statistics (no float atomics).
// Block = 256 threads = 8 waves arranged 4(M) x 2(N); each wave: 16x64 tile.
// Requires K % 64 == 0 (even number of 32-wide K-steps).
// ---------------------------------------------------------------------------
__global__ __launch_bounds__(256) void gemm_bias_stats(
    const float* __restrict__ A,       // [N][K] f32 row-major
    const __bf16* __restrict__ Wt,     // [256][K] bf16
    const float* __restrict__ bias,    // [256]
    float* __restrict__ C,             // [N][256] pre-BN output
    float* __restrict__ partial,       // [gridY][2][128][2] f32 (sum, sumsq)
    int K)
{
    // double-buffered tiles; rows padded for conflict-free ds_load_b128
    __shared__ alignas(16) float  As[2][64 * 36];   // 64 x 32 f32 (pad 36)
    __shared__ alignas(16) __bf16 Bs[2][128 * 40];  // 128 x 32 bf16 (pad 40)
    __shared__ float red[4][128][2];

    const int tid = threadIdx.x;
    const int rowBase = blockIdx.y * 64;
    const int colBase = blockIdx.x * 128;

    const int w  = tid >> 5;           // wave 0..7
    const int l  = tid & 31;
    const int wm = w >> 1;             // 0..3  (M)
    const int wn = w & 1;              // 0..1  (N)
    const int lr = l & 15;
    const int hs = l >> 4;             // half-wave select
    const int m0 = wm * 16;
    const int n0 = wn * 64;

    v8f acc[4] = {};

    // cooperative global->LDS async staging maps (per thread: 2x16B A, 2x16B B)
    const int ar = tid >> 2;           // A row 0..63
    const int ac = (tid & 3) * 8;      // A col chunk {0,8,16,24}
    const int bn = tid >> 1;           // B col 0..127
    const int bc = (tid & 1) * 16;     // B k chunk {0,16}

    const unsigned ldsA0 = LDS_OFF(&As[0][0]) + (unsigned)(ar * 144 + ac * 4);
    const unsigned ldsA1 = LDS_OFF(&As[1][0]) + (unsigned)(ar * 144 + ac * 4);
    const unsigned ldsB0 = LDS_OFF(&Bs[0][0]) + (unsigned)(bn * 80 + bc * 2);
    const unsigned ldsB1 = LDS_OFF(&Bs[1][0]) + (unsigned)(bn * 80 + bc * 2);

    // running global byte addresses; +128B (A) / +64B (B) per 32-wide K-step
    unsigned long long ga = (unsigned long long)(const void*)
        (A + (size_t)(rowBase + ar) * K + ac);
    unsigned long long gb = (unsigned long long)(const void*)
        (Wt + (size_t)(colBase + bn) * K + bc);

#define ISSUE_TILE(la, lb)                                                       \
    do {                                                                         \
        asm volatile("global_load_async_to_lds_b128 %0, %1, off"                 \
                     :: "v"(la), "v"(ga) : "memory");                            \
        asm volatile("global_load_async_to_lds_b128 %0, %1, off"                 \
                     :: "v"((la) + 16u), "v"(ga + 16ull) : "memory");            \
        asm volatile("global_load_async_to_lds_b128 %0, %1, off"                 \
                     :: "v"(lb), "v"(gb) : "memory");                            \
        asm volatile("global_load_async_to_lds_b128 %0, %1, off"                 \
                     :: "v"((lb) + 16u), "v"(gb + 16ull) : "memory");            \
        ga += 128ull; gb += 64ull;                                               \
    } while (0)

    // one 16x64-per-wave tile step: 1 A fragment, 4 B fragments, 4 WMMAs
    auto compute_tile = [&](const float* Ab, const __bf16* Bb) {
        v16bf af;
        {
            const float* p = Ab + (m0 + lr) * 36 + hs * 16;
            #pragma unroll
            for (int i = 0; i < 8; ++i) af[i] = (__bf16)p[i];
            #pragma unroll
            for (int i = 0; i < 8; ++i) af[8 + i] = (__bf16)p[8 + i];
        }
        v16bf bf0, bf1, bf2, bf3;
        {
            const __bf16* q0 = Bb + (n0 + 0 * 16 + lr) * 40 + hs * 16;
            const __bf16* q1 = Bb + (n0 + 1 * 16 + lr) * 40 + hs * 16;
            const __bf16* q2 = Bb + (n0 + 2 * 16 + lr) * 40 + hs * 16;
            const __bf16* q3 = Bb + (n0 + 3 * 16 + lr) * 40 + hs * 16;
            #pragma unroll
            for (int i = 0; i < 16; ++i) { bf0[i] = q0[i]; bf1[i] = q1[i];
                                           bf2[i] = q2[i]; bf3[i] = q3[i]; }
        }
        acc[0] = __builtin_amdgcn_wmma_f32_16x16x32_bf16(false, af, false, bf0,
                                                         (short)0, acc[0], false, false);
        acc[1] = __builtin_amdgcn_wmma_f32_16x16x32_bf16(false, af, false, bf1,
                                                         (short)0, acc[1], false, false);
        acc[2] = __builtin_amdgcn_wmma_f32_16x16x32_bf16(false, af, false, bf2,
                                                         (short)0, acc[2], false, false);
        acc[3] = __builtin_amdgcn_wmma_f32_16x16x32_bf16(false, af, false, bf3,
                                                         (short)0, acc[3], false, false);
    };

    const int nk = K >> 5;             // K-steps of 32; nk is even (K % 64 == 0)
    ISSUE_TILE(ldsA0, ldsB0);          // prologue: tile 0 -> buffer 0

    // steady state: two tiles (buffers 0,1) per trip, straight-line interior
    for (int kt = 0; kt < nk; kt += 2) {
        // ---- buffer 0 holds tile kt; prefetch tile kt+1 -> buffer 1 ----
        ISSUE_TILE(ldsA1, ldsB1);
        // async loads complete in order: <=4 outstanding => tile kt landed
        asm volatile("s_wait_asynccnt 0x4" ::: "memory");
        __syncthreads();
        compute_tile(&As[0][0], &Bs[0][0]);
        __syncthreads();               // all waves done with buffer 0

        // ---- buffer 1 holds tile kt+1; prefetch tile kt+2 -> buffer 0 ----
        if (kt + 2 < nk) {
            ISSUE_TILE(ldsA0, ldsB0);
            asm volatile("s_wait_asynccnt 0x4" ::: "memory");
        } else {
            asm volatile("s_wait_asynccnt 0x0" ::: "memory");
        }
        __syncthreads();
        compute_tile(&As[1][0], &Bs[1][0]);
        __syncthreads();               // all waves done with buffer 1
    }
#undef ISSUE_TILE

    // Epilogue: bias add, store pre-BN tensor, deterministic column stats
    #pragma unroll
    for (int t = 0; t < 4; ++t) {
        const int cl  = n0 + t * 16 + lr;        // block-local column
        const int col = colBase + cl;
        const float bv = bias[col];
        float s = 0.f, q = 0.f;
        #pragma unroll
        for (int r = 0; r < 8; ++r) {
            float v = acc[t][r] + bv;            // lane row = m0 + hs*8 + r
            C[(size_t)(rowBase + m0 + hs * 8 + r) * C_OUT + col] = v;
            s += v;
            q += v * v;
        }
        s += __shfl_down(s, 16);                 // fold M 8..15 into lanes 0..15
        q += __shfl_down(q, 16);
        if (l < 16) {
            red[wm][cl][0] = s;
            red[wm][cl][1] = q;
        }
    }
    __syncthreads();
    {
        const int cl   = tid >> 1;
        const int kind = tid & 1;
        float v = red[0][cl][kind] + red[1][cl][kind]
                + red[2][cl][kind] + red[3][cl][kind];
        partial[(((size_t)blockIdx.y * gridDim.x + blockIdx.x) * 128 + cl) * 2 + kind] = v;
    }
}

// ---------------------------------------------------------------------------
// Fold the per-workgroup partials (fixed order => deterministic) into
// per-channel BN affine: y = x*scale + shift, then relu at the consumer.
// bn[0..255] = scale, bn[256..511] = shift.
// ---------------------------------------------------------------------------
__global__ __launch_bounds__(256) void bn_finalize(
    const float* __restrict__ partial, int nWgM, float invN,
    const float* __restrict__ gamma, const float* __restrict__ beta,
    float* __restrict__ bn)
{
    int c = threadIdx.x;                 // channel 0..255
    int wgN = c >> 7, col = c & 127;
    float s = 0.f, q = 0.f;
    for (int m = 0; m < nWgM; ++m) {
        const float* p = partial + (((size_t)m * 2 + wgN) * 128 + col) * 2;
        s += p[0];
        q += p[1];
    }
    float mean  = s * invN;
    float var   = fmaxf(q * invN - mean * mean, 0.f);   // biased var (jnp.var)
    float scale = gamma[c] * rsqrtf(var + 1e-5f);
    bn[c]         = scale;
    bn[C_OUT + c] = beta[c] - mean * scale;
}

// ---------------------------------------------------------------------------
// out[i] = relu(bn_p(h_pre[cluster[i]])) + relu(bn_s(s_pre[i]))
// float4 per thread; h_pre (64 MB) lives in the 192 MB L2 so the gather is cheap.
// ---------------------------------------------------------------------------
__global__ __launch_bounds__(256) void unpool_out_k(
    const float* __restrict__ h, const float* __restrict__ s,
    const int* __restrict__ cluster,
    const float* __restrict__ bnp, const float* __restrict__ bns,
    float* __restrict__ out, int Nout)
{
    int t   = blockIdx.x * 256 + threadIdx.x;   // group of 4 channels
    int row = t >> 6;
    if (row >= Nout) return;
    int c4 = (t & 63) << 2;
    int j  = cluster[row];
    const float4 hv  = *(const float4*)(h + (size_t)j   * C_OUT + c4);
    const float4 sv  = *(const float4*)(s + (size_t)row * C_OUT + c4);
    const float4 sp  = *(const float4*)(bnp + c4);
    const float4 tp  = *(const float4*)(bnp + C_OUT + c4);
    const float4 ssc = *(const float4*)(bns + c4);
    const float4 ts  = *(const float4*)(bns + C_OUT + c4);
    float4 o;
    o.x = fmaxf(fmaf(hv.x, sp.x, tp.x), 0.f) + fmaxf(fmaf(sv.x, ssc.x, ts.x), 0.f);
    o.y = fmaxf(fmaf(hv.y, sp.y, tp.y), 0.f) + fmaxf(fmaf(sv.y, ssc.y, ts.y), 0.f);
    o.z = fmaxf(fmaf(hv.z, sp.z, tp.z), 0.f) + fmaxf(fmaf(sv.z, ssc.z, ts.z), 0.f);
    o.w = fmaxf(fmaf(hv.w, sp.w, tp.w), 0.f) + fmaxf(fmaf(sv.w, ssc.w, ts.w), 0.f);
    *(float4*)(out + (size_t)t * 4) = o;
}

// ---------------------------------------------------------------------------
extern "C" void kernel_launch(void* const* d_in, const int* in_sizes, int n_in,
                              void* d_out, int out_size, void* d_ws, size_t ws_size,
                              hipStream_t stream) {
    const float* feat    = (const float*)d_in[0];
    const float* skipf   = (const float*)d_in[1];
    const int*   cluster = (const int*)  d_in[2];
    const float* W_proj  = (const float*)d_in[3];
    const float* b_proj  = (const float*)d_in[4];
    const float* gamma_p = (const float*)d_in[5];
    const float* beta_p  = (const float*)d_in[6];
    const float* W_skip  = (const float*)d_in[7];
    const float* b_skip  = (const float*)d_in[8];
    const float* gamma_s = (const float*)d_in[9];
    const float* beta_s  = (const float*)d_in[10];
    float* out = (float*)d_out;

    const int C_IN = 512, C_SK = 256;
    const int N_in  = in_sizes[0] / C_IN;    // 65536
    const int N_out = in_sizes[2];           // 262144

    char* ws = (char*)d_ws;
    size_t off = 0;
    float*  h_pre = (float*)(ws + off);  off += (size_t)N_in  * C_OUT * 4;
    float*  s_pre = (float*)(ws + off);  off += (size_t)N_out * C_OUT * 4;
    __bf16* Wt_p  = (__bf16*)(ws + off); off += (size_t)C_OUT * C_IN * 2;
    __bf16* Wt_s  = (__bf16*)(ws + off); off += (size_t)C_OUT * C_SK * 2;
    const int nWgM1 = N_in  / 64;            // 1024
    const int nWgM2 = N_out / 64;            // 4096
    float* part1 = (float*)(ws + off);   off += (size_t)nWgM1 * 2 * 128 * 2 * 4;
    float* part2 = (float*)(ws + off);   off += (size_t)nWgM2 * 2 * 128 * 2 * 4;
    float* bnp   = (float*)(ws + off);   off += 2 * C_OUT * 4;
    float* bns   = (float*)(ws + off);   off += 2 * C_OUT * 4;

    // 1) weights -> bf16, transposed
    convert_wt<<<(C_IN * C_OUT + 255) / 256, 256, 0, stream>>>(W_proj, Wt_p, C_IN, C_IN * C_OUT);
    convert_wt<<<(C_SK * C_OUT + 255) / 256, 256, 0, stream>>>(W_skip, Wt_s, C_SK, C_SK * C_OUT);

    // 2) GEMMs (bf16 WMMA, f32 accumulate, async double-buffered staging)
    gemm_bias_stats<<<dim3(2, nWgM1), 256, 0, stream>>>(feat,  Wt_p, b_proj, h_pre, part1, C_IN);
    gemm_bias_stats<<<dim3(2, nWgM2), 256, 0, stream>>>(skipf, Wt_s, b_skip, s_pre, part2, C_SK);

    // 3) BN finalize -> per-channel scale/shift
    bn_finalize<<<1, 256, 0, stream>>>(part1, nWgM1, 1.0f / (float)N_in,  gamma_p, beta_p, bnp);
    bn_finalize<<<1, 256, 0, stream>>>(part2, nWgM2, 1.0f / (float)N_out, gamma_s, beta_s, bns);

    // 4) gather + affine + relu + add
    unpool_out_k<<<(N_out * 64 + 255) / 256, 256, 0, stream>>>(
        h_pre, s_pre, cluster, bnp, bns, out, N_out);
}
